// CIN_63350767616439
// MI455X (gfx1250) — matrix operand order repaired and checked by reading, not compiled, not run on hardware
//
#include <hip/hip_runtime.h>

// ---------------------------------------------------------------------------
// DCN-v2 cross network on MI455X (gfx1250, wave32, WMMA).
//
// Per row m = b*32+d:  out[m,o] = relu( sum_k z[m,k] * W[k,o] + bias[o] ),
// where z[m, f*Hp+hh] = x0[b,f,d] * h[b,hh,d]  (rank-1, built on the fly).
//
// GEMM: M=32768, N=256, K in {1521,4992,4992}; f16 A/B, f32 accumulate via
// v_wmma_f32_16x16x32_f16. One workgroup = 32 rows x 256 cols (one batch b).
// ---------------------------------------------------------------------------

typedef __attribute__((ext_vector_type(16))) _Float16 v16h;
typedef __attribute__((ext_vector_type(8)))  float    v8f;

// ---------------------------------------------------------------------------
// Pre-shuffle W (K x 256, f32) into per-lane B-fragment order, f16.
// B-fragment layout (16x16x32 f16 WMMA, wave32), dual of ISA 7.12.2 A layout
// (and matching the sparse 64x16 B table): lane l: col = l&15,
// K = (l>>4)*8 + (e&7) + (e>>3)*16 for vector element e in [0,16).
// Wf index = ((chunk*16 + colblock)*32 + lane)*16 + e, chunk = k/32.
// Out-of-range k (K padding) -> 0.
// ---------------------------------------------------------------------------
__global__ void prep_w_kernel(const float* __restrict__ W,
                              _Float16* __restrict__ Wf,
                              int K, int total) {
  int idx = blockIdx.x * blockDim.x + threadIdx.x;
  if (idx >= total) return;
  int e     = idx & 15;
  int lane  = (idx >> 4) & 31;
  int cb    = (idx >> 9) & 15;
  int chunk = idx >> 13;
  int k   = chunk * 32 + ((lane >> 4) << 3) + (e & 7) + ((e >> 3) << 4);
  int col = (cb << 4) + (lane & 15);
  float v = (k < K) ? W[k * 256 + col] : 0.0f;
  Wf[idx] = (_Float16)v;
}

// ---------------------------------------------------------------------------
// One cross layer. HP = 39 (h is x itself) or 128 (h from previous layer).
// Block: 256 threads = 8 waves; 32 rows (= one batch element) x 256 cols.
// Wave w: col blocks {w, w+8} -> cols [16w,16w+16) and [128+16w, 128+16w+16);
// row subtiles {0,1}; 4 v8f accumulators. B register-double-buffered.
// ---------------------------------------------------------------------------
template <int HP, bool HAS_NEXT>
__global__ __launch_bounds__(256)
void cross_layer_kernel(const float* __restrict__ x,      // (1024, 39, 32)
                        const float* __restrict__ hsrc,   // HP==39: x layout; else (M,128)
                        const _Float16* __restrict__ Wf,  // pre-shuffled
                        const float* __restrict__ bias,   // (256,)
                        const float* __restrict__ fcw,    // (512,)
                        float* __restrict__ h_next,       // (M,128) or nullptr
                        float* __restrict__ partial_out,  // (1024,) per-b partial
                        int fc_off) {
  constexpr int F    = 39;
  constexpr int XS   = 44;                       // padded x row stride (bank-safe)
  constexpr int HS   = (HP == 128) ? 132 : 44;   // padded h row stride (bank-safe)
  constexpr int K    = F * HP;
  constexpr int KPAD = ((K + 31) / 32) * 32;
  constexpr int NCH  = KPAD / 32;                // K chunks of 32

  __shared__ float x_lds[32 * XS];
  __shared__ float h_lds[32 * HS];
  __shared__ float wred[8];

  const int tid  = threadIdx.x;
  const int lane = tid & 31;
  const int w    = tid >> 5;
  const int m0   = blockIdx.x << 5;   // 32 rows per block => b = blockIdx.x
  const int bb   = blockIdx.x;

  // Stage x tile (d-inner => contiguous 128B global reads per 32 lanes).
  for (int i = tid; i < 32 * XS; i += 256) {
    int d = i & 31, c = i >> 5;
    x_lds[d * XS + c] = (c < F) ? x[(bb * F + c) * 32 + d] : 0.0f;
  }
  // Stage h tile.
  if constexpr (HP == 39) {
    for (int i = tid; i < 32 * HS; i += 256) {
      int d = i & 31, c = i >> 5;
      h_lds[d * HS + c] = (c < F) ? hsrc[(bb * F + c) * 32 + d] : 0.0f;
    }
  } else {
    for (int i = tid; i < 32 * HS; i += 256) {  // c-inner: (M,128) is contiguous
      int r = i / HS, c = i % HS;
      h_lds[i] = (c < HP) ? hsrc[(size_t)(m0 + r) * HP + c] : 0.0f;
    }
  }
  __syncthreads();

  v8f acc00 = {}, acc01 = {}, acc10 = {}, acc11 = {};

  const int half16 = lane >> 4;        // 0: K-group low, 1: K-group high
  const int abase  = half16 * 8;       // A-frag K offset per ISA 7.12.2
  const int r_in   = lane & 15;        // row within 16-row subtile / col within block
  const int cb0 = w, cb1 = w + 8;

  // B pointers: chunk stride = 16 colblocks * 32 lanes * 16 halves = 8192.
  const _Float16* wp0 = Wf + (size_t)(cb0 * 32 + lane) * 16;
  const _Float16* wp1 = Wf + (size_t)(cb1 * 32 + lane) * 16;
  v16h bc0 = *(const v16h*)wp0;
  v16h bc1 = *(const v16h*)wp1;

  for (int ch = 0; ch < NCH; ++ch) {
    const int k0 = ch * 32;
    // Prefetch next chunk's B fragments (redundant re-load on last chunk).
    const int nch = (ch + 1 < NCH) ? (ch + 1) : ch;
    const v16h bn0 = *(const v16h*)(wp0 + (size_t)nch * 8192);
    const v16h bn1 = *(const v16h*)(wp1 + (size_t)nch * 8192);

    // A fragments (16-bit A 16x32 layout, ISA 7.12.2):
    //   lanes 0-15 : elems 0..7 -> K=k0+e   ; elems 8..15 -> K=k0+16+(e-8)
    //   lanes 16-31: elems 0..7 -> K=k0+8+e ; elems 8..15 -> K=k0+24+(e-8)
    v16h afrag[2];
#pragma unroll
    for (int s = 0; s < 2; ++s) {
      const float* xr = &x_lds[(s * 16 + r_in) * XS];
      const float* hr = &h_lds[(s * 16 + r_in) * HS];
      v16h a;
      if constexpr (HP == 128) {
        // f constant across the whole 32-K chunk; hh = shifts/masks only.
        const int   f   = k0 >> 7;
        const int   hh0 = (k0 & 127) + abase;
        const float xv  = xr[f];
#pragma unroll
        for (int e = 0; e < 8; ++e) {
          a[e]     = (_Float16)(xv * hr[hh0 + e]);
          a[e + 8] = (_Float16)(xv * hr[hh0 + 16 + e]);
        }
      } else {
        // Generic path (HP=39): constant div/mod -> magic multiply.
#pragma unroll
        for (int e = 0; e < 16; ++e) {
          const int k = k0 + abase + ((e < 8) ? e : (8 + e));
          const unsigned f  = (unsigned)k / (unsigned)HP;
          const unsigned hh = (unsigned)k - f * (unsigned)HP;
          a[e] = (_Float16)(xr[f] * hr[hh]);   // k>=K => xr[f]==0 (padded)
        }
      }
      afrag[s] = a;
    }

    acc00 = __builtin_amdgcn_wmma_f32_16x16x32_f16(false, afrag[0], false, bc0,
                                                   (short)0, acc00, false, false);
    acc01 = __builtin_amdgcn_wmma_f32_16x16x32_f16(false, afrag[0], false, bc1,
                                                   (short)0, acc01, false, false);
    acc10 = __builtin_amdgcn_wmma_f32_16x16x32_f16(false, afrag[1], false, bc0,
                                                   (short)0, acc10, false, false);
    acc11 = __builtin_amdgcn_wmma_f32_16x16x32_f16(false, afrag[1], false, bc1,
                                                   (short)0, acc11, false, false);
    bc0 = bn0;
    bc1 = bn1;
  }

  // Epilogue. Per-accumulator behaviour is UNIFORM (cb0 -> cols<128,
  // cb1 -> cols>=128), so store/contribute flags are compile-time constants:
  // no divergent branches.
  float part = 0.0f;
  auto epi = [&](const v8f& acc, int s, int cb, bool doStore, bool doContrib) {
    const int   col = (cb << 4) + r_in;
    const float bv  = bias[col];
    const float wv  = doContrib ? fcw[col + fc_off] : 0.0f;
#pragma unroll
    for (int j = 0; j < 8; ++j) {
      const int m = m0 + s * 16 + j + half16 * 8;   // C/D layout, ISA 7.12.2
      float v = acc[j] + bv;
      v = v > 0.0f ? v : 0.0f;
      if (doStore) h_next[m * 128 + col] = v;
      part += v * wv;
    }
  };
  epi(acc00, 0, cb0, HAS_NEXT, !HAS_NEXT);
  epi(acc01, 0, cb1, false,    true);
  epi(acc10, 1, cb0, HAS_NEXT, !HAS_NEXT);
  epi(acc11, 1, cb1, false,    true);

#pragma unroll
  for (int off = 16; off > 0; off >>= 1)
    part += __shfl_xor(part, off, 32);
  if (lane == 0) wred[w] = part;
  __syncthreads();
  if (tid == 0) {
    float s = 0.0f;
#pragma unroll
    for (int i = 0; i < 8; ++i) s += wred[i];   // fixed order => deterministic
    partial_out[blockIdx.x] = s;
  }
}

__global__ void finalize_kernel(const float* __restrict__ p,
                                const float* __restrict__ fc_b,
                                float* __restrict__ y) {
  int b = blockIdx.x * blockDim.x + threadIdx.x;
  if (b < 1024) y[b] = fc_b[0] + p[b] + p[1024 + b] + p[2048 + b];
}

// ---------------------------------------------------------------------------
// Host-side launch. Workspace layout (bytes):
//   h1  @ 0         (32768*128*4 = 16,777,216)
//   h2  @ 16777216  (16,777,216)
//   Wf0 @ 33554432  (48 chunks  * 8192 halves * 2 =   786,432)
//   Wf1 @ 34340864  (156 chunks * 8192 * 2        = 2,555,904)
//   Wf2 @ 36896768  (2,555,904)
//   partials @ 39452672 (3*1024*4)
// ---------------------------------------------------------------------------
extern "C" void kernel_launch(void* const* d_in, const int* in_sizes, int n_in,
                              void* d_out, int out_size, void* d_ws, size_t ws_size,
                              hipStream_t stream) {
  const float* x    = (const float*)d_in[0];
  const float* W0   = (const float*)d_in[1];
  const float* b0   = (const float*)d_in[2];
  const float* W1   = (const float*)d_in[3];
  const float* b1   = (const float*)d_in[4];
  const float* W2   = (const float*)d_in[5];
  const float* b2   = (const float*)d_in[6];
  const float* fc_w = (const float*)d_in[7];
  const float* fc_b = (const float*)d_in[8];
  float* y = (float*)d_out;

  char* ws = (char*)d_ws;
  float*    h1   = (float*)(ws + 0);
  float*    h2   = (float*)(ws + 16777216);
  _Float16* Wf0  = (_Float16*)(ws + 33554432);
  _Float16* Wf1  = (_Float16*)(ws + 34340864);
  _Float16* Wf2  = (_Float16*)(ws + 36896768);
  float*    part = (float*)(ws + 39452672);

  // Pre-shuffle weights into B-fragment order (f16).
  const int n0 = 48 * 8192, n12 = 156 * 8192;
  prep_w_kernel<<<(n0  + 255) / 256, 256, 0, stream>>>(W0, Wf0, 1521, n0);
  prep_w_kernel<<<(n12 + 255) / 256, 256, 0, stream>>>(W1, Wf1, 4992, n12);
  prep_w_kernel<<<(n12 + 255) / 256, 256, 0, stream>>>(W2, Wf2, 4992, n12);

  // Cross layers (1024 blocks = one batch element each).
  cross_layer_kernel<39, true ><<<1024, 256, 0, stream>>>(x, x,  Wf0, b0, fc_w, h1,      part + 0,    -128);
  cross_layer_kernel<128, true ><<<1024, 256, 0, stream>>>(x, h1, Wf1, b1, fc_w, h2,      part + 1024,    0);
  cross_layer_kernel<128, false><<<1024, 256, 0, stream>>>(x, h2, Wf2, b2, fc_w, nullptr, part + 2048,  256);

  finalize_kernel<<<4, 256, 0, stream>>>(part, fc_b, y);
}
